// CausalSelfAttention_19018115186956
// MI455X (gfx1250) — matrix operand order repaired and checked by reading, not compiled, not run on hardware
//
#include <hip/hip_runtime.h>

typedef __attribute__((ext_vector_type(16))) _Float16 v16h;
typedef __attribute__((ext_vector_type(8)))  float    v8f;
typedef unsigned int u32x4 __attribute__((ext_vector_type(4)));
typedef int          i32x8 __attribute__((ext_vector_type(8)));
typedef int          i32x4 __attribute__((ext_vector_type(4)));

#define D_MODEL  1024
#define N_HEADS  16
#define HEAD_DIM 64
#define T_SEQ    2048
#define B_BATCH  4
#define M_TOT    (B_BATCH * T_SEQ)   // 8192

// ---- lane-pair exchange (xor 16) via ds_swizzle, wave32 ----
static __device__ __forceinline__ int swz16i(int v) {
    return __builtin_amdgcn_ds_swizzle(v, 0x401F);  // BITMASK_PERM: xor=0x10, and=0x1f
}
static __device__ __forceinline__ float swz16f(float v) {
    union { float f; int i; } u; u.f = v; u.i = swz16i(u.i); return u.f;
}

// ---- TDM: 2-D tile (16-bit elements) global -> LDS, with row padding ----
// D# per CDNA5 ISA ch.8: group0 = count/lds_addr/global_addr/type,
// group1 = data_size + pad cfg + tensor/tile dims + strides. Groups 2/3 zero (2-D).
static __device__ __forceinline__ void tdm_load_2d_f16(
    const void* gptr,              // global tile start
    unsigned int lds_byte_off,     // LDS dest byte offset
    unsigned int row_len,          // tensor_dim0 (elements)
    unsigned long long row_stride, // tensor_dim0_stride (elements)
    unsigned int tile_w,           // tile_dim0 (elements per row)
    unsigned int tile_h,           // tile_dim1 (rows)
    unsigned int pad_ival_code,    // pad after 2^(code+1) DWORDs
    unsigned int pad_amt_code)     // pad of (code+1) DWORDs
{
    unsigned long long ga = (unsigned long long)(uintptr_t)gptr;
    u32x4 g0;
    g0[0] = 1u;                                    // count=1, is_restore=0, gather=0
    g0[1] = lds_byte_off;                          // lds_addr [63:32]
    g0[2] = (unsigned int)ga;                      // global_addr [95:64]
    g0[3] = (unsigned int)((ga >> 32) & 0x01FFFFFFu) | (2u << 30);  // addr hi | type=2
    i32x8 g1;
    g1[0] = (int)((1u << 16) |                     // data_size = 1 -> 2 bytes
                  (1u << 20) |                     // pad_enable
                  (pad_ival_code << 22) |
                  (pad_amt_code  << 25));
    g1[1] = (int)((row_len & 0xFFFFu) << 16);      // tensor_dim0[15:0] @ word1[31:16]
    g1[2] = (int)(((row_len >> 16) & 0xFFFFu) |    // tensor_dim0[31:16]
                  (0xFFFFu << 16));                // tensor_dim1[15:0] = 65535 (no OOB)
    g1[3] = (int)((tile_w & 0xFFFFu) << 16);       // tensor_dim1 hi=0 | tile_dim0
    g1[4] = (int)(tile_h & 0xFFFFu);               // tile_dim1 | tile_dim2=0
    g1[5] = (int)(unsigned int)row_stride;         // tensor_dim0_stride[31:0]
    g1[6] = (int)(unsigned int)((row_stride >> 32) & 0xFFFFu); // stride[47:32] | dim1_stride lo
    g1[7] = 0;
    i32x4 z4 = (i32x4)0;
    i32x8 z8 = (i32x8)0;
    __builtin_amdgcn_tensor_load_to_lds(g0, g1, z4, z4, z8, 0);
}

static __device__ __forceinline__ unsigned int lds_offset_of(const void* p) {
    // flat address of a __shared__ object: addr[31:0] is the LDS byte offset (ISA 10.2)
    return (unsigned int)(uintptr_t)p;
}

// ---------------- conversion kernels ----------------
__global__ __launch_bounds__(256) void convert_f32_to_f16(const float* __restrict__ src,
                                                          _Float16* __restrict__ dst, int n) {
    int i = (blockIdx.x * 256 + threadIdx.x) * 4;
    if (i < n) {
        float4 v = *(const float4*)(src + i);
        union { _Float16 h[4]; uint2 u; } o;
        o.h[0] = (_Float16)v.x; o.h[1] = (_Float16)v.y;
        o.h[2] = (_Float16)v.z; o.h[3] = (_Float16)v.w;
        *(uint2*)(dst + i) = o.u;
    }
}

// wt[n][k] = (f16) w[k][n] ; K == 1024, one block per n, 256 threads x 4 k each
__global__ __launch_bounds__(256) void transpose_w_to_f16(const float* __restrict__ w,
                                                          _Float16* __restrict__ wt,
                                                          int K, int N) {
    int n = blockIdx.x;
    int k = threadIdx.x * 4;
    union { _Float16 h[4]; uint2 u; } o;
#pragma unroll
    for (int j = 0; j < 4; ++j) o.h[j] = (_Float16)w[(size_t)(k + j) * N + n];
    *(uint2*)(wt + (size_t)n * K + k) = o.u;
}

// ---------------- WMMA GEMM:  C[M][N] = A[M][K] * Bt[N][K]^T + bias ----------------
// Double-buffered TDM pipeline: DMA of tile i+1 overlaps the 8 WMMAs per wave of tile i.
// MODE 0: QKV gemm -> scatter to Qh/Kh (head layout, f16) and Vt (transposed, f16)
// MODE 1: proj gemm -> f32 output
template <int MODE>
__global__ __launch_bounds__(256) void gemm_wmma(
    const _Float16* __restrict__ A, const _Float16* __restrict__ Bt,
    const float* __restrict__ bias,
    _Float16* __restrict__ Qh, _Float16* __restrict__ Kh, _Float16* __restrict__ Vt,
    float* __restrict__ Out, int N, int K)
{
    __shared__ _Float16 As[2][128 * 40];   // ping-pong, rows TDM-padded 32->40 halves
    __shared__ _Float16 Bs[2][128 * 40];

    const int tid  = threadIdx.x;
    const int lane = tid & 31;
    const int wave = tid >> 5;
    const int mBase = blockIdx.y * 128;
    const int nBase = blockIdx.x * 128;
    const int wm = (wave >> 1) * 32;    // wave M offset: 0/32/64/96
    const int wn = (wave & 1) * 64;     // wave N offset: 0/64

    const int lr = lane & 15;
    const int hi = lane >> 4;           // half-wave id
    const int aOff0 = hi ? 8 : 0;       // A-frag K offsets: {0,16} or {8,24}

    const unsigned int asOff[2] = { lds_offset_of(&As[0][0]), lds_offset_of(&As[1][0]) };
    const unsigned int bsOff[2] = { lds_offset_of(&Bs[0][0]), lds_offset_of(&Bs[1][0]) };

    v8f acc[2][4];
#pragma unroll
    for (int i = 0; i < 2; ++i)
#pragma unroll
        for (int j = 0; j < 4; ++j) acc[i][j] = (v8f)0.f;

    const int NT = K / 32;
    // prologue: DMA tile 0 into buffer 0
    if (wave == 0)      tdm_load_2d_f16(A  + (size_t)mBase * K, asOff[0], 32,
                                        (unsigned long long)K, 32, 128, 3, 3);
    else if (wave == 1) tdm_load_2d_f16(Bt + (size_t)nBase * K, bsOff[0], 32,
                                        (unsigned long long)K, 32, 128, 3, 3);

    for (int it = 0; it < NT; ++it) {
        const int cur = it & 1;
        const int k0 = it * 32;
        if (it + 1 < NT) {
            // issue next tile into the other buffer, then wait for the oldest
            if (wave == 0)      tdm_load_2d_f16(A  + (size_t)mBase * K + k0 + 32, asOff[cur ^ 1],
                                                32, (unsigned long long)K, 32, 128, 3, 3);
            else if (wave == 1) tdm_load_2d_f16(Bt + (size_t)nBase * K + k0 + 32, bsOff[cur ^ 1],
                                                32, (unsigned long long)K, 32, 128, 3, 3);
            if (wave < 2) __builtin_amdgcn_s_wait_tensorcnt(1);
        } else {
            if (wave < 2) __builtin_amdgcn_s_wait_tensorcnt(0);
        }
        __syncthreads();

        v16h af[2], bf[4];
#pragma unroll
        for (int ms = 0; ms < 2; ++ms) {
            const _Float16* p = &As[cur][(wm + ms * 16 + lr) * 40];
            union { v16h v; uint4 u[2]; } r;
            r.u[0] = *(const uint4*)(p + aOff0);
            r.u[1] = *(const uint4*)(p + aOff0 + 16);
            af[ms] = r.v;
        }
#pragma unroll
        for (int ns = 0; ns < 4; ++ns) {
            const _Float16* p = &Bs[cur][(wn + ns * 16 + lr) * 40 + hi * 8];
            union { v16h v; uint4 u[2]; } r;
            r.u[0] = *(const uint4*)(p);
            r.u[1] = *(const uint4*)(p + 16);
            bf[ns] = r.v;
        }
#pragma unroll
        for (int ms = 0; ms < 2; ++ms)
#pragma unroll
            for (int ns = 0; ns < 4; ++ns)
                acc[ms][ns] = __builtin_amdgcn_wmma_f32_16x16x32_f16(
                    false, af[ms], false, bf[ns], (short)0, acc[ms][ns], false, false);
        __syncthreads();   // protects buffer reuse two iterations later
    }

    // epilogue (C-layout: lane<16 -> col=lane, rows r; lane>=16 -> col=lane-16, rows 8+r)
#pragma unroll
    for (int ms = 0; ms < 2; ++ms) {
#pragma unroll
        for (int ns = 0; ns < 4; ++ns) {
            int n = nBase + wn + ns * 16 + lr;
            float bv = bias[n];
#pragma unroll
            for (int r = 0; r < 8; ++r) {
                int m = mBase + wm + ms * 16 + r + hi * 8;
                float v = acc[ms][ns][r] + bv;
                if (MODE == 1) {
                    Out[(size_t)m * N + n] = v;
                } else {
                    int b = m >> 11, t = m & 2047;
                    int which = n >> 10;          // 0=Q 1=K 2=V
                    int n2 = n & 1023;
                    int h = n2 >> 6, d = n2 & 63;
                    int bh = b * N_HEADS + h;
                    _Float16 hv = (_Float16)v;
                    if (which == 0)      Qh[((size_t)bh * T_SEQ + t) * HEAD_DIM + d] = hv;
                    else if (which == 1) Kh[((size_t)bh * T_SEQ + t) * HEAD_DIM + d] = hv;
                    else                 Vt[((size_t)bh * HEAD_DIM + d) * T_SEQ + t] = hv;
                }
            }
        }
    }
}

// ---------------- flash attention (wave32, S^T = K*Q^T trick, double-buffered TDM) ----------------
__global__ __launch_bounds__(256) void attn_kernel(
    const _Float16* __restrict__ Qh, const _Float16* __restrict__ Kh,
    const _Float16* __restrict__ Vt, _Float16* __restrict__ Yatt)
{
    __shared__ _Float16 Ks[2][64 * 72];   // [t][hd], TDM-padded 64->72, ping-pong
    __shared__ _Float16 Vs[2][64 * 72];   // [d][t],  TDM-padded (V pre-transposed in global)

    const int tid  = threadIdx.x;
    const int lane = tid & 31;
    const int wave = tid >> 5;
    const int bh = blockIdx.y;
    const int b  = bh >> 4, h = bh & 15;
    const int q0 = blockIdx.x * 128 + wave * 16;
    const int lr = lane & 15;
    const int hi = lane >> 4;
    const int q  = q0 + lr;
    const int aOff0 = hi ? 8 : 0;

    const unsigned int ksOff[2] = { lds_offset_of(&Ks[0][0]), lds_offset_of(&Ks[1][0]) };
    const unsigned int vsOff[2] = { lds_offset_of(&Vs[0][0]), lds_offset_of(&Vs[1][0]) };

    const _Float16* kBase = Kh + (size_t)bh * T_SEQ * HEAD_DIM;
    const _Float16* vBase = Vt + (size_t)bh * HEAD_DIM * T_SEQ;

    // Q^T B-fragments (two hd-chunks of 32) -- straight from global, hd contiguous
    v16h qb[2];
    {
        const _Float16* qp = Qh + ((size_t)bh * T_SEQ + q) * HEAD_DIM + hi * 8;
#pragma unroll
        for (int c = 0; c < 2; ++c) {
            union { v16h v; uint4 u[2]; } r;
            r.u[0] = *(const uint4*)(qp + c * 32);
            r.u[1] = *(const uint4*)(qp + c * 32 + 16);
            qb[c] = r.v;
        }
    }

    float m_i = -3.0e38f, l_i = 0.f;
    v8f od[4];
#pragma unroll
    for (int i = 0; i < 4; ++i) od[i] = (v8f)0.f;

    const int NKT = blockIdx.x * 2 + 2;   // causal bound: tiles of 64 up to q-block end
    // prologue: DMA tile 0 into buffer 0
    if (wave == 0)      tdm_load_2d_f16(kBase, ksOff[0], 64, 64ull, 64, 64, 4, 3);
    else if (wave == 1) tdm_load_2d_f16(vBase, vsOff[0], 64, (unsigned long long)T_SEQ,
                                        64, 64, 4, 3);

    for (int it = 0; it < NKT; ++it) {
        const int cur = it & 1;
        const int kt = it * 64;
        if (it + 1 < NKT) {
            if (wave == 0)      tdm_load_2d_f16(kBase + (size_t)(kt + 64) * HEAD_DIM,
                                                ksOff[cur ^ 1], 64, 64ull, 64, 64, 4, 3);
            else if (wave == 1) tdm_load_2d_f16(vBase + kt + 64, vsOff[cur ^ 1],
                                                64, (unsigned long long)T_SEQ, 64, 64, 4, 3);
            if (wave < 2) __builtin_amdgcn_s_wait_tensorcnt(1);
        } else {
            if (wave < 2) __builtin_amdgcn_s_wait_tensorcnt(0);
        }
        __syncthreads();

        if (kt <= q0) {   // wave-uniform: skip fully-masked tiles
            // S^T = K * Q^T : 4 subtiles over t (lane=q col, regs=t rows)
            v8f st[4];
#pragma unroll
            for (int sub = 0; sub < 4; ++sub) {
                st[sub] = (v8f)0.f;
                const _Float16* kp = &Ks[cur][(sub * 16 + lr) * 72];
#pragma unroll
                for (int c = 0; c < 2; ++c) {
                    union { v16h v; uint4 u[2]; } r;
                    r.u[0] = *(const uint4*)(kp + c * 32 + aOff0);
                    r.u[1] = *(const uint4*)(kp + c * 32 + aOff0 + 16);
                    st[sub] = __builtin_amdgcn_wmma_f32_16x16x32_f16(
                        false, r.v, false, qb[c], (short)0, st[sub], false, false);
                }
            }
            // scale + causal mask + online softmax (q is the lane dim!)
            float p[4][8];
            float mloc = -3.0e38f;
#pragma unroll
            for (int sub = 0; sub < 4; ++sub)
#pragma unroll
                for (int r = 0; r < 8; ++r) {
                    int t = kt + sub * 16 + r + hi * 8;
                    float v = st[sub][r] * 0.125f;           // 1/sqrt(64)
                    v = (t > q) ? -3.0e38f : v;
                    p[sub][r] = v;
                    mloc = fmaxf(mloc, v);
                }
            mloc = fmaxf(mloc, swz16f(mloc));                // combine lane & lane^16 (same q)
            float mnew  = fmaxf(m_i, mloc);
            float scale = __expf(m_i - mnew);
            float ls = 0.f;
#pragma unroll
            for (int sub = 0; sub < 4; ++sub)
#pragma unroll
                for (int r = 0; r < 8; ++r) {
                    float e = __expf(p[sub][r] - mnew);
                    p[sub][r] = e;
                    ls += e;
                }
            ls += swz16f(ls);
            l_i = l_i * scale + ls;
            m_i = mnew;
#pragma unroll
            for (int i = 0; i < 4; ++i)
#pragma unroll
                for (int r = 0; r < 8; ++r) od[i][r] *= scale;

            // pack P (f32->f16) : pk[sub][j] holds this lane's 8 probs of subtile sub
            uint32_t pk[4][4];
#pragma unroll
            for (int sub = 0; sub < 4; ++sub)
#pragma unroll
                for (int j = 0; j < 4; ++j) {
                    union { _Float16 hh[2]; uint32_t u; } w;
                    w.hh[0] = (_Float16)p[sub][2 * j];
                    w.hh[1] = (_Float16)p[sub][2 * j + 1];
                    pk[sub][j] = w.u;
                }
            // build P^T B-fragments via xor-16 swizzles, then O^T += V^T * P^T
#pragma unroll
            for (int c = 0; c < 2; ++c) {
                union { v16h v; uint32_t u[8]; } bf;
#pragma unroll
                for (int j = 0; j < 4; ++j) {
                    uint32_t lo  = pk[2 * c][j];
                    uint32_t hih = pk[2 * c + 1][j];
                    uint32_t swl = (uint32_t)swz16i((int)lo);
                    uint32_t swh = (uint32_t)swz16i((int)hih);
                    bf.u[j]     = hi ? swh : lo;   // halves 0..7
                    bf.u[4 + j] = hi ? hih : swl;  // halves 8..15
                }
#pragma unroll
                for (int dsub = 0; dsub < 4; ++dsub) {
                    const _Float16* vp = &Vs[cur][(dsub * 16 + lr) * 72 + c * 32];
                    union { v16h v; uint4 u[2]; } av;
                    av.u[0] = *(const uint4*)(vp + hi * 16);
                    av.u[1] = *(const uint4*)(vp + hi * 16 + 8);
                    od[dsub] = __builtin_amdgcn_wmma_f32_16x16x32_f16(
                        false, av.v, false, bf.v, (short)0, od[dsub], false, false);
                }
            }
        }
        __syncthreads();   // protects buffer reuse two iterations later
    }

    // epilogue: O^T layout -> lane = q, regs = d ; one b128 store per d-subtile
    float inv = 1.f / l_i;
#pragma unroll
    for (int dsub = 0; dsub < 4; ++dsub) {
        union { uint4 u; _Float16 hv[8]; } o;
#pragma unroll
        for (int r = 0; r < 8; ++r) o.hv[r] = (_Float16)(od[dsub][r] * inv);
        size_t off = ((size_t)(b * T_SEQ + q)) * D_MODEL + h * HEAD_DIM + dsub * 16 + hi * 8;
        *(uint4*)(Yatt + off) = o.u;
    }
}

// ---------------- host launch ----------------
extern "C" void kernel_launch(void* const* d_in, const int* in_sizes, int n_in,
                              void* d_out, int out_size, void* d_ws, size_t ws_size,
                              hipStream_t stream) {
    const float* x      = (const float*)d_in[0];
    const float* w_qkv  = (const float*)d_in[1];
    const float* b_qkv  = (const float*)d_in[2];
    const float* w_proj = (const float*)d_in[3];
    const float* b_proj = (const float*)d_in[4];
    float* out = (float*)d_out;

    _Float16* ws = (_Float16*)d_ws;
    size_t o = 0;
    _Float16* xh     = ws + o; o += (size_t)M_TOT * D_MODEL;        // 16 MB
    _Float16* wqkvT  = ws + o; o += (size_t)3 * D_MODEL * D_MODEL;  //  6 MB
    _Float16* wprojT = ws + o; o += (size_t)D_MODEL * D_MODEL;      //  2 MB
    _Float16* Qh     = ws + o; o += (size_t)M_TOT * D_MODEL;        // 16 MB
    _Float16* Kh     = ws + o; o += (size_t)M_TOT * D_MODEL;        // 16 MB
    _Float16* Vt     = ws + o; o += (size_t)M_TOT * D_MODEL;        // 16 MB
    _Float16* Yatt   = ws + o; o += (size_t)M_TOT * D_MODEL;        // 16 MB

    convert_f32_to_f16<<<(M_TOT * D_MODEL / 4 + 255) / 256, 256, 0, stream>>>(
        x, xh, M_TOT * D_MODEL);
    transpose_w_to_f16<<<3 * D_MODEL, 256, 0, stream>>>(w_qkv, wqkvT, D_MODEL, 3 * D_MODEL);
    transpose_w_to_f16<<<D_MODEL, 256, 0, stream>>>(w_proj, wprojT, D_MODEL, D_MODEL);

    dim3 g1(3 * D_MODEL / 128, M_TOT / 128);
    gemm_wmma<0><<<g1, 256, 0, stream>>>(xh, wqkvT, b_qkv, Qh, Kh, Vt, nullptr,
                                         3 * D_MODEL, D_MODEL);

    dim3 g2(T_SEQ / 128, B_BATCH * N_HEADS);
    attn_kernel<<<g2, 256, 0, stream>>>(Qh, Kh, Vt, Yatt);

    dim3 g3(D_MODEL / 128, M_TOT / 128);
    gemm_wmma<1><<<g3, 256, 0, stream>>>(Yatt, wprojT, b_proj, nullptr, nullptr, nullptr,
                                         out, D_MODEL, D_MODEL);
}
